// Model_12386685681918
// MI455X (gfx1250) — matrix operand order repaired and checked by reading, not compiled, not run on hardware
//
#include <hip/hip_runtime.h>
#include <hip/hip_bf16.h>

// ---------------------------------------------------------------------------
// Problem constants (match the JAX reference)
// ---------------------------------------------------------------------------
#define Bb   2
#define Nn   1024
#define Cf   64
#define Ss   512
#define Kk   32
#define Dm   128
#define Hh   4
#define Dh   32          // Dm / Hh
#define CH   67          // Cf + 3
#define DIN  131         // 2*Cf + 3
#define DINP 160         // DIN padded up to multiple of 32 for WMMA K-loop

typedef __attribute__((ext_vector_type(16))) _Float16 v16h;
typedef __attribute__((ext_vector_type(8)))  float    v8f;

// CDNA5 wave32 WMMA 16x16x32 f16 fragment maps (ISA 7.12.2):
//  A (16x32 MxK): lane L row m=L&15; halves h[0..7]  = K 8*hi+0..7,
//                                      h[8..15] = K 16+8*hi+0..7   (hi=L>>4)
//  B (32x16 KxN): lane L col n=L&15; half p = same K as A (this code loads both A and B as K p+16*hi)
//  C/D (16x16 f32): vgpr j, lane L -> m = j + 8*hi, n = L&15
union F16frag { v16h v; _Float16 h[16]; };
union F32acc  { v8f  v; float    f[8]; };

// ---------------------------------------------------------------------------
// Async global->LDS staging (gfx1250 ASYNCcnt path), with safe fallback.
// Builtin signature (from hipcc diagnostic): arg0 = <4 x i32> addrspace(1)*,
// arg1 = <4 x i32> addrspace(3)*, then two int immediates (offset, cpol).
// ---------------------------------------------------------------------------
typedef __attribute__((__vector_size__(16))) int v4i32;

__device__ __forceinline__ void async_copy_b128(const void* g, void* l) {
#if defined(__gfx1250__) && __has_builtin(__builtin_amdgcn_global_load_async_to_lds_b128)
    __builtin_amdgcn_global_load_async_to_lds_b128(
        (__attribute__((address_space(1))) v4i32*)(void*)g,
        (__attribute__((address_space(3))) v4i32*)l, 0, 0);
#else
    *(uint4*)l = *(const uint4*)g;
#endif
}
__device__ __forceinline__ void async_copy_wait() {
#if defined(__gfx1250__) && __has_builtin(__builtin_amdgcn_global_load_async_to_lds_b128)
  #if __has_builtin(__builtin_amdgcn_s_wait_asynccnt)
    __builtin_amdgcn_s_wait_asynccnt(0);
  #else
    asm volatile("s_wait_asynccnt 0" ::: "memory");
  #endif
#endif
}

// ---------------------------------------------------------------------------
// 1) Furthest point sampling: one block per batch, 1024 threads (1 per point)
// ---------------------------------------------------------------------------
__global__ void fps_kernel(const float* __restrict__ xyz,
                           int* __restrict__ fps_idx,
                           float* __restrict__ new_xyz) {
    __shared__ float sdist[1024];
    __shared__ int   sidx[1024];
    __shared__ float cx, cy, cz;
    __shared__ int   curFar;
    const int b = blockIdx.x, t = threadIdx.x;
    const float px = xyz[(size_t)(b * Nn + t) * 3 + 0];
    const float py = xyz[(size_t)(b * Nn + t) * 3 + 1];
    const float pz = xyz[(size_t)(b * Nn + t) * 3 + 2];
    float dist = 1e10f;
    if (t == 0) curFar = 0;
    __syncthreads();
    for (int s = 0; s < Ss; ++s) {
        const int far = curFar;
        if (t == 0) {
            fps_idx[b * Ss + s] = far;
            cx = xyz[(size_t)(b * Nn + far) * 3 + 0];
            cy = xyz[(size_t)(b * Nn + far) * 3 + 1];
            cz = xyz[(size_t)(b * Nn + far) * 3 + 2];
            new_xyz[(size_t)(b * Ss + s) * 3 + 0] = cx;
            new_xyz[(size_t)(b * Ss + s) * 3 + 1] = cy;
            new_xyz[(size_t)(b * Ss + s) * 3 + 2] = cz;
        }
        __syncthreads();
        const float dx = px - cx, dy = py - cy, dz = pz - cz;
        dist = fminf(dist, dx * dx + dy * dy + dz * dz);
        sdist[t] = dist; sidx[t] = t;
        __syncthreads();
        for (int off = 512; off > 0; off >>= 1) {
            if (t < off) {
                if (sdist[t + off] > sdist[t]) {   // strict > keeps first-occurrence argmax
                    sdist[t] = sdist[t + off]; sidx[t] = sidx[t + off];
                }
            }
            __syncthreads();
        }
        if (t == 0) curFar = sidx[0];
        __syncthreads();
    }
}

// ---------------------------------------------------------------------------
// 2) KNN-32: one block per (b,s) query; distances in LDS, 32 successive argmins
// ---------------------------------------------------------------------------
__global__ void knn32_kernel(const float* __restrict__ xyz,
                             const float* __restrict__ new_xyz,
                             int* __restrict__ knn_idx) {
    __shared__ float d[Nn];
    __shared__ float rb[128];
    __shared__ int   ri[128];
    const int b = blockIdx.x / Ss, s = blockIdx.x % Ss, t = threadIdx.x;
    const float qx = new_xyz[(size_t)(b * Ss + s) * 3 + 0];
    const float qy = new_xyz[(size_t)(b * Ss + s) * 3 + 1];
    const float qz = new_xyz[(size_t)(b * Ss + s) * 3 + 2];
    for (int i = t; i < Nn; i += 128) {
        const float dx = qx - xyz[(size_t)(b * Nn + i) * 3 + 0];
        const float dy = qy - xyz[(size_t)(b * Nn + i) * 3 + 1];
        const float dz = qz - xyz[(size_t)(b * Nn + i) * 3 + 2];
        d[i] = dx * dx + dy * dy + dz * dz;
    }
    __syncthreads();
    for (int kk = 0; kk < Kk; ++kk) {
        float best = 3e38f; int bi = Nn;
        for (int i = t; i < Nn; i += 128)
            if (d[i] < best) { best = d[i]; bi = i; }
        rb[t] = best; ri[t] = bi;
        __syncthreads();
        for (int off = 64; off > 0; off >>= 1) {
            if (t < off) {
                if (rb[t + off] < rb[t] ||
                    (rb[t + off] == rb[t] && ri[t + off] < ri[t])) {
                    rb[t] = rb[t + off]; ri[t] = ri[t + off];
                }
            }
            __syncthreads();
        }
        if (t == 0) { knn_idx[((size_t)(b * Ss + s)) * Kk + kk] = ri[0]; d[ri[0]] = 3e38f; }
        __syncthreads();
    }
}

// ---------------------------------------------------------------------------
// 3) Per-group mean over K, and per-batch sum of squared centered values
// ---------------------------------------------------------------------------
__global__ void zero_f32_kernel(float* p, int n) {
    int i = blockIdx.x * blockDim.x + threadIdx.x;
    if (i < n) p[i] = 0.0f;
}

__global__ void group_stats_kernel(const float* __restrict__ xyz,
                                   const float* __restrict__ feats,
                                   const int* __restrict__ knn_idx,
                                   float* __restrict__ meanb,
                                   float* __restrict__ ssq) {
    const int b = blockIdx.x / Ss, s = blockIdx.x % Ss, c = threadIdx.x;
    if (c >= CH) return;
    float vals[Kk];
    float sum = 0.0f;
    for (int k = 0; k < Kk; ++k) {
        const int idx = knn_idx[((size_t)(b * Ss + s)) * Kk + k];
        const float v = (c < Cf) ? feats[(size_t)(b * Nn + idx) * Cf + c]
                                 : xyz[(size_t)(b * Nn + idx) * 3 + (c - Cf)];
        vals[k] = v; sum += v;
    }
    const float mean = sum * (1.0f / (float)Kk);
    meanb[((size_t)(b * Ss + s)) * CH + c] = mean;
    float sq = 0.0f;
    for (int k = 0; k < Kk; ++k) { const float t = vals[k] - mean; sq += t * t; }
    atomicAdd(&ssq[b], sq);
}

// ---------------------------------------------------------------------------
// 4) Normalize + affine + concat anchor feats -> f16 [B*S*K, DINP] (zero pad)
// ---------------------------------------------------------------------------
__global__ void build_points_kernel(const float* __restrict__ xyz,
                                    const float* __restrict__ feats,
                                    const int* __restrict__ knn_idx,
                                    const int* __restrict__ fps_idx,
                                    const float* __restrict__ meanb,
                                    const float* __restrict__ ssq,
                                    const float* __restrict__ alpha,
                                    const float* __restrict__ beta,
                                    _Float16* __restrict__ np16) {
    const int row = blockIdx.x;               // b*S*K + s*K + k
    const int c = threadIdx.x;                // 0..159
    const int b = row / (Ss * Kk);
    const int s = (row / Kk) % Ss;
    const int k = row % Kk;
    float out = 0.0f;
    if (c < CH) {
        const int idx = knn_idx[((size_t)(b * Ss + s)) * Kk + k];
        const float v = (c < Cf) ? feats[(size_t)(b * Nn + idx) * Cf + c]
                                 : xyz[(size_t)(b * Nn + idx) * 3 + (c - Cf)];
        const float mean = meanb[((size_t)(b * Ss + s)) * CH + c];
        const float std  = sqrtf(ssq[b] / (float)(Ss * Kk * CH - 1));
        out = alpha[c] * ((v - mean) / (std + 1e-5f)) + beta[c];
    } else if (c < DIN) {
        const int a = fps_idx[b * Ss + s];
        out = feats[(size_t)(b * Nn + a) * Cf + (c - CH)];
    }
    np16[(size_t)row * DINP + c] = (_Float16)out;
}

// ---------------------------------------------------------------------------
// 5) Weight conversion: write f16 weights PRE-TRANSPOSED  W^T[n][k]
//    so the WMMA B-fragment (fixed n, contiguous k) is a pure vector load.
// ---------------------------------------------------------------------------
__global__ void cvtT_f16_kernel(const float* __restrict__ src,   // [K, Nc] row-major
                                _Float16* __restrict__ dst,      // [Nc, K]
                                int K, int Nc) {
    int i = blockIdx.x * blockDim.x + threadIdx.x;
    if (i < K * Nc) {
        const int n = i / K, k = i % K;
        dst[i] = (_Float16)src[(size_t)k * Nc + n];
    }
}
__global__ void padT_wt_kernel(const float* __restrict__ src,    // [DIN, Dm]
                               _Float16* __restrict__ dst) {     // [Dm, DINP]
    int i = blockIdx.x * blockDim.x + threadIdx.x;
    if (i < Dm * DINP) {
        const int n = i / DINP, k = i % DINP;
        dst[i] = (k < DIN) ? (_Float16)src[(size_t)k * Dm + n] : (_Float16)0.0f;
    }
}

// ---------------------------------------------------------------------------
// 6) WMMA GEMM: C[M,N] = A[M,Kc](f16, row-major) * BT[N,Kc](f16, transposed)
//    block = 128 threads (4 waves), 32x32 output tile.
//    Tiles staged via async global->LDS b128; all fragment loads are uint4.
//    M, N, Kc multiples of 32.  Optional transposed f16 output (for vh^T).
// ---------------------------------------------------------------------------
__global__ void gemm16_kernel(const _Float16* __restrict__ A, int lda,
                              const _Float16* __restrict__ BT, int ldbT,
                              int M, int N, int Kc,
                              const float* __restrict__ bias,
                              const float* __restrict__ bng,
                              const float* __restrict__ bnb,
                              int relu,
                              float* __restrict__ outF32,
                              _Float16* __restrict__ outF16,
                              _Float16* __restrict__ outF16T) {
    __shared__ _Float16 As [32][40];   // A rows (m)   x k-slab, 80B row stride
    __shared__ _Float16 BsT[32][40];   // B^T rows (n) x k-slab
    const int tid = threadIdx.x;
    const int lane = tid & 31, wave = tid >> 5;
    const int wm = wave >> 1, wn = wave & 1;
    const int l15 = lane & 15, hi = lane >> 4;
    const int m0 = blockIdx.x * 32, n0 = blockIdx.y * 32;
    const int r = tid >> 2, seg = tid & 3;

    v8f acc = {};
    for (int kt = 0; kt < Kc; kt += 32) {
        async_copy_b128(&A [(size_t)(m0 + r) * lda  + kt + seg * 8], &As [r][seg * 8]);
        async_copy_b128(&BT[(size_t)(n0 + r) * ldbT + kt + seg * 8], &BsT[r][seg * 8]);
        async_copy_wait();
        __syncthreads();
        F16frag af, bf;
        const _Float16* ap = &As [16 * wm + l15][0];
        ((uint4*)af.h)[0] = *(const uint4*)(ap + 16 * hi);
        ((uint4*)af.h)[1] = *(const uint4*)(ap + 16 * hi + 8);
        const _Float16* bp = &BsT[16 * wn + l15][0];
        ((uint4*)bf.h)[0] = *(const uint4*)(bp + 16 * hi);
        ((uint4*)bf.h)[1] = *(const uint4*)(bp + 16 * hi + 8);
        acc = __builtin_amdgcn_wmma_f32_16x16x32_f16(
            false, af.v, false, bf.v, (short)0, acc, false, false);
        __syncthreads();
    }
    F32acc c; c.v = acc;
#pragma unroll
    for (int j = 0; j < 8; ++j) {
        const int m = m0 + 16 * wm + (j + 8 * hi);
        const int n = n0 + 16 * wn + l15;
        float y = c.f[j];
        if (bias) y += bias[n];
        if (bng)  y = y * (bng[n] * rsqrtf(1.0f + 1e-5f)) + bnb[n];
        if (relu) y = fmaxf(y, 0.0f);
        if (outF32)  outF32 [(size_t)m * N + n] = y;
        if (outF16)  outF16 [(size_t)m * N + n] = (_Float16)y;
        if (outF16T) outF16T[(size_t)n * M + m] = (_Float16)y;   // [N, M]
    }
}

// ---------------------------------------------------------------------------
// 7) Max over K (32 rows) -> v_k  (f32 for interpolation, f16 for GEMMs)
// ---------------------------------------------------------------------------
__global__ void maxk_kernel(const float* __restrict__ ybuf,
                            float* __restrict__ vk32,
                            _Float16* __restrict__ vk16) {
    const int bs = blockIdx.x, c = threadIdx.x;
    float m = -3e38f;
    for (int k = 0; k < Kk; ++k)
        m = fmaxf(m, ybuf[((size_t)bs * Kk + k) * Dm + c]);
    vk32[(size_t)bs * Dm + c] = m;
    vk16[(size_t)bs * Dm + c] = (_Float16)m;
}

// ---------------------------------------------------------------------------
// 8) 3-NN inverse-distance interpolation: one thread per (b,n)
// ---------------------------------------------------------------------------
__global__ void interp_kernel(const float* __restrict__ xyz,
                              const float* __restrict__ new_xyz,
                              const float* __restrict__ vk32,
                              _Float16* __restrict__ vq16) {
    const int g = blockIdx.x * blockDim.x + threadIdx.x;
    if (g >= Bb * Nn) return;
    const int b = g / Nn, n = g % Nn;
    const float qx = xyz[(size_t)g * 3 + 0];
    const float qy = xyz[(size_t)g * 3 + 1];
    const float qz = xyz[(size_t)g * 3 + 2];
    float d0 = 3e38f, d1 = 3e38f, d2 = 3e38f;
    int   i0 = 0, i1 = 0, i2 = 0;
    for (int s = 0; s < Ss; ++s) {
        const float dx = qx - new_xyz[(size_t)(b * Ss + s) * 3 + 0];
        const float dy = qy - new_xyz[(size_t)(b * Ss + s) * 3 + 1];
        const float dz = qz - new_xyz[(size_t)(b * Ss + s) * 3 + 2];
        const float d = dx * dx + dy * dy + dz * dz;
        if (d < d0)      { d2 = d1; i2 = i1; d1 = d0; i1 = i0; d0 = d; i0 = s; }
        else if (d < d1) { d2 = d1; i2 = i1; d1 = d; i1 = s; }
        else if (d < d2) { d2 = d; i2 = s; }
    }
    const float w0 = 1.0f / fmaxf(d0, 1e-10f);
    const float w1 = 1.0f / fmaxf(d1, 1e-10f);
    const float w2 = 1.0f / fmaxf(d2, 1e-10f);
    const float inv = 1.0f / (w0 + w1 + w2);
    const float a0 = w0 * inv, a1 = w1 * inv, a2 = w2 * inv;
    for (int c = 0; c < Dm; ++c) {
        const float v = a0 * vk32[(size_t)(b * Ss + i0) * Dm + c]
                      + a1 * vk32[(size_t)(b * Ss + i1) * Dm + c]
                      + a2 * vk32[(size_t)(b * Ss + i2) * Dm + c];
        vq16[(size_t)(b * Nn + n) * Dm + c] = (_Float16)v;
    }
}

// ---------------------------------------------------------------------------
// 9) Relative position encoding: rpe[b,h,n,s] = relu(rel@Wp1+bp1)@Wp2+bp2
//    one block per (b,n), 512 threads (one per s), weights staged in LDS.
// ---------------------------------------------------------------------------
__global__ void rpe_kernel(const float* __restrict__ xyz,
                           const float* __restrict__ new_xyz,
                           const float* __restrict__ Wp1,
                           const float* __restrict__ bp1,
                           const float* __restrict__ Wp2,
                           const float* __restrict__ bp2,
                           float* __restrict__ rpe) {
    __shared__ float sW1[3 * Dm];
    __shared__ float sb1[Dm];
    __shared__ float sW2[Dm * Hh];
    __shared__ float sb2[Hh];
    const int t = threadIdx.x;
    for (int i = t; i < 3 * Dm; i += Ss) sW1[i] = Wp1[i];
    for (int i = t; i < Dm; i += Ss)     sb1[i] = bp1[i];
    for (int i = t; i < Dm * Hh; i += Ss) sW2[i] = Wp2[i];
    if (t < Hh) sb2[t] = bp2[t];
    __syncthreads();
    const int b = blockIdx.x / Nn, n = blockIdx.x % Nn;
    const int s = t;
    const float rx = xyz[(size_t)(b * Nn + n) * 3 + 0] - new_xyz[(size_t)(b * Ss + s) * 3 + 0];
    const float ry = xyz[(size_t)(b * Nn + n) * 3 + 1] - new_xyz[(size_t)(b * Ss + s) * 3 + 1];
    const float rz = xyz[(size_t)(b * Nn + n) * 3 + 2] - new_xyz[(size_t)(b * Ss + s) * 3 + 2];
    float acc0 = sb2[0], acc1 = sb2[1], acc2 = sb2[2], acc3 = sb2[3];
    for (int tt = 0; tt < Dm; ++tt) {
        float h = fmaf(rx, sW1[tt], fmaf(ry, sW1[Dm + tt], fmaf(rz, sW1[2 * Dm + tt], sb1[tt])));
        h = fmaxf(h, 0.0f);
        acc0 = fmaf(h, sW2[tt * Hh + 0], acc0);
        acc1 = fmaf(h, sW2[tt * Hh + 1], acc1);
        acc2 = fmaf(h, sW2[tt * Hh + 2], acc2);
        acc3 = fmaf(h, sW2[tt * Hh + 3], acc3);
    }
    rpe[(size_t)((b * Hh + 0) * Nn + n) * Ss + s] = acc0;
    rpe[(size_t)((b * Hh + 1) * Nn + n) * Ss + s] = acc1;
    rpe[(size_t)((b * Hh + 2) * Nn + n) * Ss + s] = acc2;
    rpe[(size_t)((b * Hh + 3) * Nn + n) * Ss + s] = acc3;
}

// ---------------------------------------------------------------------------
// 10) Fused attention: one wave per (b, head, 16-query tile).
//     WMMA energy -> scale + RPE -> softmax (LDS, f16 probs) -> WMMA attn @ V.
//     All fragments are contiguous dual-uint4 loads (vh consumed transposed).
// ---------------------------------------------------------------------------
__global__ void attn_kernel(const _Float16* __restrict__ qh,   // [B*N, Dm]
                            const _Float16* __restrict__ kh,   // [B*S, Dm]
                            const _Float16* __restrict__ vhT,  // [Dm, B*S]
                            const float* __restrict__ rpe,     // [B,H,N,S]
                            _Float16* __restrict__ attnOut) {  // [B*N, Dm]
    __shared__ float    eT[16][Ss];    // 32 KB energies
    __shared__ _Float16 pT[16][Ss];    // 16 KB softmax probs (f16 for WMMA A)
    const int lane = threadIdx.x;
    const int bh = blockIdx.x;         // b*Hh + hd
    const int b = bh >> 2, hd = bh & 3;
    const int i0 = blockIdx.y * 16;
    const int l15 = lane & 15, hi = lane >> 4;
    const float scale = 0.17677669529663687f;   // 1/sqrt(Dh)

    F16frag aq;
    {
        const _Float16* qp = &qh[(size_t)(b * Nn + i0 + l15) * Dm + hd * Dh];
        ((uint4*)aq.h)[0] = *(const uint4*)(qp + 16 * hi);
        ((uint4*)aq.h)[1] = *(const uint4*)(qp + 16 * hi + 8);
    }

    // energy tiles: q[16x32] x k^T[32x16] per 16-key slab
    for (int jt = 0; jt < Ss / 16; ++jt) {
        F16frag bk;
        const _Float16* kp = &kh[(size_t)(b * Ss + jt * 16 + l15) * Dm + hd * Dh + 16 * hi];
        ((uint4*)bk.h)[0] = *(const uint4*)(kp);
        ((uint4*)bk.h)[1] = *(const uint4*)(kp + 8);
        v8f e = {};
        e = __builtin_amdgcn_wmma_f32_16x16x32_f16(
            false, aq.v, false, bk.v, (short)0, e, false, false);
        F32acc ec; ec.v = e;
#pragma unroll
        for (int j = 0; j < 8; ++j) {
            const int row = j + 8 * hi;
            const int col = jt * 16 + l15;
            eT[row][col] = ec.f[j] * scale +
                rpe[(size_t)((b * Hh + hd) * Nn + (i0 + row)) * Ss + col];
        }
    }
    __syncthreads();

    // softmax: lanes 0..15 each own one query row; emit f16 probabilities
    if (lane < 16) {
        float mx = -3e38f;
        for (int j = 0; j < Ss; ++j) mx = fmaxf(mx, eT[lane][j]);
        float sum = 0.0f;
        for (int j = 0; j < Ss; ++j) {
            const float ex = expf(eT[lane][j] - mx);
            eT[lane][j] = ex; sum += ex;
        }
        const float inv = 1.0f / sum;
        for (int j = 0; j < Ss; ++j) pT[lane][j] = (_Float16)(eT[lane][j] * inv);
    }
    __syncthreads();

    // out = attn[16x512] x vh[512x32] : two N-halves of Dh=32, K-step 32
    v8f o0 = {}, o1 = {};
    for (int jt = 0; jt < Ss / 32; ++jt) {
        F16frag aa, bv0, bv1;
        const _Float16* pp = &pT[l15][jt * 32];
        ((uint4*)aa.h)[0] = *(const uint4*)(pp + 16 * hi);
        ((uint4*)aa.h)[1] = *(const uint4*)(pp + 16 * hi + 8);
        const _Float16* v0p = &vhT[(size_t)(hd * Dh + l15)      * (Bb * Ss) + b * Ss + jt * 32 + 16 * hi];
        const _Float16* v1p = &vhT[(size_t)(hd * Dh + 16 + l15) * (Bb * Ss) + b * Ss + jt * 32 + 16 * hi];
        ((uint4*)bv0.h)[0] = *(const uint4*)(v0p);
        ((uint4*)bv0.h)[1] = *(const uint4*)(v0p + 8);
        ((uint4*)bv1.h)[0] = *(const uint4*)(v1p);
        ((uint4*)bv1.h)[1] = *(const uint4*)(v1p + 8);
        o0 = __builtin_amdgcn_wmma_f32_16x16x32_f16(
            false, aa.v, false, bv0.v, (short)0, o0, false, false);
        o1 = __builtin_amdgcn_wmma_f32_16x16x32_f16(
            false, aa.v, false, bv1.v, (short)0, o1, false, false);
    }
    F32acc c0, c1; c0.v = o0; c1.v = o1;
#pragma unroll
    for (int j = 0; j < 8; ++j) {
        const int m = i0 + j + 8 * hi;
        attnOut[(size_t)(b * Nn + m) * Dm + hd * Dh + l15]      = (_Float16)c0.f[j];
        attnOut[(size_t)(b * Nn + m) * Dm + hd * Dh + 16 + l15] = (_Float16)c1.f[j];
    }
}

// ---------------------------------------------------------------------------
// Host orchestration
// ---------------------------------------------------------------------------
extern "C" void kernel_launch(void* const* d_in, const int* in_sizes, int n_in,
                              void* d_out, int out_size, void* d_ws, size_t ws_size,
                              hipStream_t stream) {
    const float* xyz   = (const float*)d_in[0];
    const float* feats = (const float*)d_in[1];
    const float* alpha = (const float*)d_in[2];
    const float* beta  = (const float*)d_in[3];
    const float* W_t   = (const float*)d_in[4];
    const float* b_t   = (const float*)d_in[5];
    const float* bn_g  = (const float*)d_in[6];
    const float* bn_b  = (const float*)d_in[7];
    const float* Wq    = (const float*)d_in[8];
    const float* bq    = (const float*)d_in[9];
    const float* Wk    = (const float*)d_in[10];
    const float* bk    = (const float*)d_in[11];
    const float* Wv    = (const float*)d_in[12];
    const float* bv    = (const float*)d_in[13];
    const float* Wo    = (const float*)d_in[14];
    const float* bo    = (const float*)d_in[15];
    const float* Wp1   = (const float*)d_in[16];
    const float* bp1   = (const float*)d_in[17];
    const float* Wp2   = (const float*)d_in[18];
    const float* bp2   = (const float*)d_in[19];
    float* out = (float*)d_out;

    // Workspace carving (~47 MB total), 256B aligned slabs.
    char* wp = (char*)d_ws;
    auto carve = [&](size_t bytes) -> void* {
        void* r = (void*)wp;
        wp += (bytes + 255) & ~(size_t)255;
        return r;
    };
    int*      fps_idx = (int*)carve((size_t)Bb * Ss * 4);
    float*    new_xyz = (float*)carve((size_t)Bb * Ss * 3 * 4);
    int*      knn_idx = (int*)carve((size_t)Bb * Ss * Kk * 4);
    float*    meanb   = (float*)carve((size_t)Bb * Ss * CH * 4);
    float*    ssq     = (float*)carve((size_t)Bb * 4);
    _Float16* np16    = (_Float16*)carve((size_t)Bb * Ss * Kk * DINP * 2);
    _Float16* WtT16   = (_Float16*)carve((size_t)Dm * DINP * 2);   // [Dm][DINP]
    _Float16* WqT16   = (_Float16*)carve((size_t)Dm * Dm * 2);     // [Dm][Dm]
    _Float16* WkT16   = (_Float16*)carve((size_t)Dm * Dm * 2);
    _Float16* WvT16   = (_Float16*)carve((size_t)Dm * Dm * 2);
    _Float16* WoT16   = (_Float16*)carve((size_t)Dm * Dm * 2);
    float*    ybuf    = (float*)carve((size_t)Bb * Ss * Kk * Dm * 4);
    float*    vk32    = (float*)carve((size_t)Bb * Ss * Dm * 4);
    _Float16* vk16    = (_Float16*)carve((size_t)Bb * Ss * Dm * 2);
    _Float16* vq16    = (_Float16*)carve((size_t)Bb * Nn * Dm * 2);
    _Float16* qh16    = (_Float16*)carve((size_t)Bb * Nn * Dm * 2);
    _Float16* kh16    = (_Float16*)carve((size_t)Bb * Ss * Dm * 2);
    _Float16* vhT16   = (_Float16*)carve((size_t)Dm * Bb * Ss * 2); // [Dm][B*S]
    float*    rpeBuf  = (float*)carve((size_t)Bb * Hh * Nn * Ss * 4);
    _Float16* at16    = (_Float16*)carve((size_t)Bb * Nn * Dm * 2);

    const float* nullf = nullptr;
    float* nullf32 = nullptr;
    _Float16* nullf16 = nullptr;

    // ---- LocalGrouper ----
    fps_kernel<<<Bb, 1024, 0, stream>>>(xyz, fps_idx, new_xyz);
    knn32_kernel<<<Bb * Ss, 128, 0, stream>>>(xyz, new_xyz, knn_idx);
    zero_f32_kernel<<<1, 32, 0, stream>>>(ssq, Bb);
    group_stats_kernel<<<Bb * Ss, 96, 0, stream>>>(xyz, feats, knn_idx, meanb, ssq);
    build_points_kernel<<<Bb * Ss * Kk, DINP, 0, stream>>>(
        xyz, feats, knn_idx, fps_idx, meanb, ssq, alpha, beta, np16);

    // ---- f16 weight staging (transposed) ----
    padT_wt_kernel<<<(Dm * DINP + 255) / 256, 256, 0, stream>>>(W_t, WtT16);
    cvtT_f16_kernel<<<(Dm * Dm + 255) / 256, 256, 0, stream>>>(Wq, WqT16, Dm, Dm);
    cvtT_f16_kernel<<<(Dm * Dm + 255) / 256, 256, 0, stream>>>(Wk, WkT16, Dm, Dm);
    cvtT_f16_kernel<<<(Dm * Dm + 255) / 256, 256, 0, stream>>>(Wv, WvT16, Dm, Dm);
    cvtT_f16_kernel<<<(Dm * Dm + 255) / 256, 256, 0, stream>>>(Wo, WoT16, Dm, Dm);

    // ---- transfer conv (WMMA GEMM) + BN + ReLU, then max over K ----
    gemm16_kernel<<<dim3(Bb * Ss * Kk / 32, Dm / 32), 128, 0, stream>>>(
        np16, DINP, WtT16, DINP, Bb * Ss * Kk, Dm, DINP,
        b_t, bn_g, bn_b, 1, ybuf, nullf16, nullf16);
    maxk_kernel<<<Bb * Ss, Dm, 0, stream>>>(ybuf, vk32, vk16);

    // ---- feature propagation (3-NN interpolation) ----
    interp_kernel<<<(Bb * Nn + 255) / 256, 256, 0, stream>>>(xyz, new_xyz, vk32, vq16);

    // ---- Q/K/V projections (WMMA GEMMs); V written transposed for attention ----
    gemm16_kernel<<<dim3(Bb * Ss / 32, Dm / 32), 128, 0, stream>>>(
        vk16, Dm, WkT16, Dm, Bb * Ss, Dm, Dm, bk, nullf, nullf, 0, nullf32, kh16, nullf16);
    gemm16_kernel<<<dim3(Bb * Ss / 32, Dm / 32), 128, 0, stream>>>(
        vk16, Dm, WvT16, Dm, Bb * Ss, Dm, Dm, bv, nullf, nullf, 0, nullf32, nullf16, vhT16);
    gemm16_kernel<<<dim3(Bb * Nn / 32, Dm / 32), 128, 0, stream>>>(
        vq16, Dm, WqT16, Dm, Bb * Nn, Dm, Dm, bq, nullf, nullf, 0, nullf32, qh16, nullf16);

    // ---- relative position encoding (VALU MLP, weights in LDS) ----
    rpe_kernel<<<Bb * Nn, Ss, 0, stream>>>(xyz, new_xyz, Wp1, bp1, Wp2, bp2, rpeBuf);

    // ---- fused WMMA attention ----
    attn_kernel<<<dim3(Bb * Hh, Nn / 16), 32, 0, stream>>>(qh16, kh16, vhT16, rpeBuf, at16);

    // ---- output projection (WMMA GEMM) -> d_out ----
    gemm16_kernel<<<dim3(Bb * Nn / 32, Dm / 32), 128, 0, stream>>>(
        at16, Dm, WoT16, Dm, Bb * Nn, Dm, Dm, bo, nullf, nullf, 0, out, nullf16, nullf16);
}
